// MultiScaleAttention_33646773797048
// MI455X (gfx1250) — matrix operand order repaired and checked by reading, not compiled, not run on hardware
//
#include <hip/hip_runtime.h>
#include <hip/hip_bf16.h>
#include <cstdint>

// ---------------------------------------------------------------------------
// MultiScaleAttention (MViT pooling attention) for gfx1250 / MI455X.
// Heavy math: v_wmma_f32_16x16x32_bf16 (wave32 WMMA).
// Data movement: global_load_async_to_lds_b128 (ASYNCcnt) double-buffered.
// ---------------------------------------------------------------------------

typedef __attribute__((ext_vector_type(16))) __bf16 bf16x16;
typedef __attribute__((ext_vector_type(8)))  float   f32x8;

union U8 { uint4 u; __bf16 h[8]; };

// Build a 16-element bf16 A/B fragment from two contiguous 16B chunks.
__device__ __forceinline__ bf16x16 ldfrag(const __bf16* p0, const __bf16* p1) {
  U8 lo, hi;
  lo.u = *(const uint4*)p0;
  hi.u = *(const uint4*)p1;
  bf16x16 f;
#pragma unroll
  for (int i = 0; i < 8; ++i) { f[i] = lo.h[i]; f[i + 8] = hi.h[i]; }
  return f;
}

__device__ __forceinline__ f32x8 wmma_bf16(bf16x16 a, bf16x16 b, f32x8 c) {
  return __builtin_amdgcn_wmma_f32_16x16x32_bf16(false, a, false, b,
                                                 (short)0, c, false, false);
}

// Async DMA: copy 16B from global memory directly into LDS (no VGPR data).
// Tracked by ASYNCcnt; generic LDS pointer's low 32 bits are the LDS offset.
__device__ __forceinline__ void async_copy16(const void* gptr, void* lptr) {
  uint32_t loff = (uint32_t)(uintptr_t)lptr;
  asm volatile("global_load_async_to_lds_b128 %0, %1, off"
               :: "v"(loff), "v"(gptr) : "memory");
}

__device__ __forceinline__ void wait_async0() {
  asm volatile("s_wait_asynccnt 0" ::: "memory");
}

// ---------------------------------------------------------------------------
// fp32 -> bf16 bulk conversion (one-shot; feeds async-DMA GEMM staging).
// ---------------------------------------------------------------------------
__global__ __launch_bounds__(256)
void cvt_bf16(const float* __restrict__ s, __bf16* __restrict__ d, long n) {
  long i = ((long)blockIdx.x * 256 + threadIdx.x) * 8;
  if (i >= n) return;
  if (i + 8 <= n) {
    float4 a = *(const float4*)(s + i);
    float4 b = *(const float4*)(s + i + 4);
    U8 o;
    o.h[0] = (__bf16)a.x; o.h[1] = (__bf16)a.y;
    o.h[2] = (__bf16)a.z; o.h[3] = (__bf16)a.w;
    o.h[4] = (__bf16)b.x; o.h[5] = (__bf16)b.y;
    o.h[6] = (__bf16)b.z; o.h[7] = (__bf16)b.w;
    *(uint4*)(d + i) = o.u;
  } else {
    for (long j = i; j < n; ++j) d[j] = (__bf16)s[j];
  }
}

// ---------------------------------------------------------------------------
// GEMM: D[M,N] = A[M,K]bf16 * W[N,K]bf16^T (+bias), D fp32.
// Block = 256 thr (8 waves), tile 128x128, wave = 32x64 (2x4 WMMA accs).
// K staged 32-wide via async-to-LDS, double-buffered (DMA overlaps WMMA).
// Tail M rows are clamped (their outputs are masked at the store).
// ---------------------------------------------------------------------------
__global__ __launch_bounds__(256)
void gemm_bf16_wmma(const __bf16* __restrict__ A, const __bf16* __restrict__ W,
                    const float* __restrict__ bias, float* __restrict__ D,
                    int M, int N, int K) {
  __shared__ __attribute__((aligned(16))) __bf16 lA[2][128 * 32];
  __shared__ __attribute__((aligned(16))) __bf16 lW[2][128 * 32];

  const int t    = threadIdx.x;
  const int lane = t & 31;
  const int w    = t >> 5;
  const int l16  = lane & 15;
  const int half = lane >> 4;
  const int bm   = blockIdx.x * 128;
  const int bn   = blockIdx.y * 128;
  const int wm   = (w & 3) * 32;   // wave M offset in tile
  const int wn   = (w >> 2) * 64;  // wave N offset in tile

  f32x8 acc[2][4];
#pragma unroll
  for (int i = 0; i < 2; ++i)
#pragma unroll
    for (int j = 0; j < 4; ++j) acc[i][j] = (f32x8)0.0f;

  // Stage one 128x32 bf16 tile pair: 512 x 16B chunks each, 2 per thread.
  auto stage = [&](int k0, int buf) {
#pragma unroll
    for (int p = 0; p < 2; ++p) {
      int chunk = t + 256 * p;        // 0..511
      int row   = chunk >> 2;         // 0..127
      int off   = (chunk & 3) * 8;    // bf16 offset 0,8,16,24
      int gm = bm + row; if (gm > M - 1) gm = M - 1;   // clamp tail rows
      async_copy16(A + (size_t)gm * K + k0 + off, &lA[buf][row * 32 + off]);
      int gn = bn + row;              // N,K are multiples of 128/32: in range
      async_copy16(W + (size_t)gn * K + k0 + off, &lW[buf][row * 32 + off]);
    }
  };

  stage(0, 0);
  wait_async0();
  __syncthreads();

  for (int k0 = 0; k0 < K; k0 += 32) {
    const int buf = (k0 >> 5) & 1;
    if (k0 + 32 < K) stage(k0 + 32, buf ^ 1);   // DMA next tile during compute

    // A fragment (16x32): lane m=l16; k = {half*8+0..7, half*8+16..23}
    bf16x16 af[2];
#pragma unroll
    for (int mt = 0; mt < 2; ++mt) {
      int m = wm + mt * 16 + l16;
      af[mt] = ldfrag(&lA[buf][m * 32 + half * 8],
                      &lA[buf][m * 32 + half * 8 + 16]);
    }
    // B fragment (32x16): lane n=l16; k = {half*8+0..7, half*8+16..23}
    bf16x16 wf[4];
#pragma unroll
    for (int nt = 0; nt < 4; ++nt) {
      int n = wn + nt * 16 + l16;
      wf[nt] = ldfrag(&lW[buf][n * 32 + half * 8],
                      &lW[buf][n * 32 + half * 8 + 16]);
    }
#pragma unroll
    for (int mt = 0; mt < 2; ++mt)
#pragma unroll
      for (int nt = 0; nt < 4; ++nt)
        acc[mt][nt] = wmma_bf16(af[mt], wf[nt], acc[mt][nt]);

    wait_async0();       // next tile landed (own wave) ...
    __syncthreads();     // ... and all waves' DMAs/reads are done
  }

  // Epilogue: C/D layout: VGPR r -> row r+8*half, col l16.
#pragma unroll
  for (int mt = 0; mt < 2; ++mt) {
#pragma unroll
    for (int nt = 0; nt < 4; ++nt) {
      int col  = bn + wn + nt * 16 + l16;
      float bv = bias ? bias[col] : 0.0f;
#pragma unroll
      for (int r = 0; r < 8; ++r) {
        int row = bm + wm + mt * 16 + r + 8 * half;
        if (row < M) D[(size_t)row * N + col] = acc[mt][nt][r] + bv;
      }
    }
  }
}

// ---------------------------------------------------------------------------
// Depthwise 3x3x3 pooling conv (stride 1,s,s) + cls passthrough + layernorm
// over hd=64. One wave per output token; lane owns channels d and d+32.
// Input: proj output fp32 (B, 6273, 768); output bf16 (B, 12, Np, 64).
// ---------------------------------------------------------------------------
__global__ __launch_bounds__(128)
void pool_norm(const float* __restrict__ src, const float* __restrict__ pw,
               const float* __restrict__ g, const float* __restrict__ beta,
               __bf16* __restrict__ dst, int stride, int OH, int OW) {
  const int Np   = 1 + 8 * OH * OW;
  const int wave = threadIdx.x >> 5;
  const int lane = threadIdx.x & 31;
  long token = (long)blockIdx.x * 4 + wave;
  long total = (long)8 * 12 * Np;
  if (token >= total) return;                 // uniform per wave
  int tok = (int)(token % Np);
  int bh  = (int)(token / Np);
  int h   = bh % 12;
  int b   = bh / 12;

  float v[2];
#pragma unroll
  for (int j = 0; j < 2; ++j) {
    int d = lane + j * 32;
    float acc;
    if (tok == 0) {
      acc = src[(size_t)b * 6273 * 768 + (size_t)h * 64 + d];
    } else {
      int sp = tok - 1;
      int ow = sp % OW;
      int oh = (sp / OW) % OH;
      int ot = sp / (OW * OH);
      acc = 0.0f;
#pragma unroll
      for (int kt = 0; kt < 3; ++kt) {
#pragma unroll
        for (int kh = 0; kh < 3; ++kh) {
#pragma unroll
          for (int kw = 0; kw < 3; ++kw) {
            int it = ot + kt - 1;
            int ih = oh * stride + kh - 1;
            int iw = ow * stride + kw - 1;
            if (it >= 0 && it < 8 && ih >= 0 && ih < 28 && iw >= 0 && iw < 28) {
              int itok = 1 + (it * 28 + ih) * 28 + iw;
              acc += pw[((d * 3 + kt) * 3 + kh) * 3 + kw] *
                     src[((size_t)b * 6273 + itok) * 768 + h * 64 + d];
            }
          }
        }
      }
    }
    v[j] = acc;
  }
  // mean / var over 64 channels (2 per lane, 32 lanes)
  float s = v[0] + v[1];
#pragma unroll
  for (int m = 1; m < 32; m <<= 1) s += __shfl_xor(s, m, 32);
  float mu = s * (1.0f / 64.0f);
  float d0 = v[0] - mu, d1 = v[1] - mu;
  float q = d0 * d0 + d1 * d1;
#pragma unroll
  for (int m = 1; m < 32; m <<= 1) q += __shfl_xor(q, m, 32);
  float inv = rsqrtf(q * (1.0f / 64.0f) + 1e-5f);
#pragma unroll
  for (int j = 0; j < 2; ++j) {
    int d = lane + j * 32;
    float y = (v[j] - mu) * inv * g[d] + beta[d];
    dst[(((size_t)b * 12 + h) * Np + tok) * 64 + d] = (__bf16)y;
  }
}

// ---------------------------------------------------------------------------
// Fused attention: softmax(q*scale @ k^T) @ v, flash-style online softmax.
// Block = 128 thr (4 waves); each wave owns 16 q rows (64 per block).
// K staged via async-to-LDS; V^T staged with a VALU transpose (~110 KB LDS).
// Output bf16 at (b*1569+q, h*64+d) feeding the projection GEMM directly.
// ---------------------------------------------------------------------------
#define ATT_NQ  1569
#define ATT_NK  393
#define ATT_NKP 416

__global__ __launch_bounds__(128)
void attn_wmma(const __bf16* __restrict__ Q, const __bf16* __restrict__ Kb,
               const __bf16* __restrict__ Vb, __bf16* __restrict__ O,
               float scale) {
  __shared__ __attribute__((aligned(16))) __bf16 lK[ATT_NKP * 64];
  __shared__ __attribute__((aligned(16))) __bf16 lVt[64 * ATT_NKP];
  __shared__ __attribute__((aligned(16))) __bf16 stage[4][16 * 32];

  const int t    = threadIdx.x;
  const int lane = t & 31;
  const int w    = t >> 5;
  const int l16  = lane & 15;
  const int half = lane >> 4;
  const int bh   = blockIdx.y;
  const int h    = bh % 12;
  const int b    = bh / 12;

  const __bf16* kp = Kb + (size_t)bh * ATT_NK * 64;
  const __bf16* vp = Vb + (size_t)bh * ATT_NK * 64;

  // K: async DMA row-major (pad rows clamped; masked in softmax below).
  // V: VALU transpose to d-major (pad columns get p==0, contribute nothing).
  for (int i = t; i < ATT_NKP * 8; i += 128) {   // 16B chunks, 8 per row
    int n   = i >> 3;
    int off = (i & 7) * 8;
    int nc  = n < ATT_NK ? n : ATT_NK - 1;
    async_copy16(kp + (size_t)nc * 64 + off, &lK[n * 64 + off]);
    U8 vv;
    vv.u = *(const uint4*)(vp + (size_t)nc * 64 + off);
#pragma unroll
    for (int j = 0; j < 8; ++j) lVt[(off + j) * ATT_NKP + n] = vv.h[j];
  }
  wait_async0();
  __syncthreads();

  // Q fragments for this wave's 16 rows (clamped; tail rows discarded later).
  const int qbase = blockIdx.x * 64 + w * 16;
  int qrow = qbase + l16;
  if (qrow > ATT_NQ - 1) qrow = ATT_NQ - 1;
  const __bf16* qp = Q + ((size_t)bh * ATT_NQ + qrow) * 64;
  bf16x16 qf[2];
#pragma unroll
  for (int c = 0; c < 2; ++c)
    qf[c] = ldfrag(qp + c * 32 + half * 8, qp + c * 32 + half * 8 + 16);

  f32x8 o_acc[4];
#pragma unroll
  for (int i = 0; i < 4; ++i) o_acc[i] = (f32x8)0.0f;
  float rm[8], rl[8];
#pragma unroll
  for (int r = 0; r < 8; ++r) { rm[r] = -1e30f; rl[r] = 0.0f; }

  for (int jt = 0; jt < ATT_NKP / 32; ++jt) {
    // S tiles: columns [jt*32, jt*32+16) and [jt*32+16, jt*32+32)
    f32x8 s0 = (f32x8)0.0f, s1 = (f32x8)0.0f;
#pragma unroll
    for (int c = 0; c < 2; ++c) {
      int n0 = jt * 32 + l16;
      int n1 = n0 + 16;
      bf16x16 b0 = ldfrag(&lK[n0 * 64 + c * 32 + half * 8],
                          &lK[n0 * 64 + c * 32 + half * 8 + 16]);
      bf16x16 b1 = ldfrag(&lK[n1 * 64 + c * 32 + half * 8],
                          &lK[n1 * 64 + c * 32 + half * 8 + 16]);
      s0 = wmma_bf16(qf[c], b0, s0);
      s1 = wmma_bf16(qf[c], b1, s1);
    }

    // Online softmax per row r (row = r + 8*half, spread over 16 lanes).
#pragma unroll
    for (int r = 0; r < 8; ++r) {
      float a  = s0[r] * scale;
      float bs = s1[r] * scale;
      int c0 = jt * 32 + l16;
      if (c0 >= ATT_NK)      a  = -1e30f;
      if (c0 + 16 >= ATT_NK) bs = -1e30f;
      float mx = fmaxf(a, bs);
#pragma unroll
      for (int m = 1; m < 16; m <<= 1) mx = fmaxf(mx, __shfl_xor(mx, m, 32));
      float mold  = rm[r];
      float mnew  = fmaxf(mold, mx);
      float alpha = __expf(mold - mnew);
      float p0 = __expf(a - mnew);
      float p1 = __expf(bs - mnew);
      float ls = p0 + p1;
#pragma unroll
      for (int m = 1; m < 16; m <<= 1) ls += __shfl_xor(ls, m, 32);
      rl[r] = rl[r] * alpha + ls;
      rm[r] = mnew;
      o_acc[0][r] *= alpha;
      o_acc[1][r] *= alpha;
      o_acc[2][r] *= alpha;
      o_acc[3][r] *= alpha;
      int row = r + 8 * half;
      stage[w][row * 32 + l16]      = (__bf16)p0;   // C-layout -> LDS
      stage[w][row * 32 + 16 + l16] = (__bf16)p1;
    }

    // Reload P as an A fragment (per-wave stage; in-wave DS ordering).
    bf16x16 pf = ldfrag(&stage[w][l16 * 32 + half * 8],
                        &stage[w][l16 * 32 + half * 8 + 16]);
#pragma unroll
    for (int nt = 0; nt < 4; ++nt) {
      int col = nt * 16 + l16;
      bf16x16 vf = ldfrag(&lVt[col * ATT_NKP + jt * 32 + half * 8],
                          &lVt[col * ATT_NKP + jt * 32 + half * 8 + 16]);
      o_acc[nt] = wmma_bf16(pf, vf, o_acc[nt]);
    }
  }

  // Finalize: O /= rowsum, write bf16 at (b*Nq + q, h*64 + d).
#pragma unroll
  for (int r = 0; r < 8; ++r) {
    float inv = 1.0f / rl[r];
    int row = qbase + r + 8 * half;
    if (row < ATT_NQ) {
      size_t base = ((size_t)b * ATT_NQ + row) * 768 + (size_t)h * 64;
#pragma unroll
      for (int nt = 0; nt < 4; ++nt)
        O[base + nt * 16 + l16] = (__bf16)(o_acc[nt][r] * inv);
    }
  }
}

// ---------------------------------------------------------------------------
// Host launcher.
// ---------------------------------------------------------------------------
extern "C" void kernel_launch(void* const* d_in, const int* in_sizes, int n_in,
                              void* d_out, int out_size, void* d_ws,
                              size_t ws_size, hipStream_t stream) {
  const float* x      = (const float*)d_in[0];
  const float* Wq     = (const float*)d_in[1];
  const float* Wk     = (const float*)d_in[2];
  const float* Wv     = (const float*)d_in[3];
  const float* Wproj  = (const float*)d_in[4];
  const float* b_proj = (const float*)d_in[5];
  const float* pqw    = (const float*)d_in[6];
  const float* pkw    = (const float*)d_in[7];
  const float* pvw    = (const float*)d_in[8];
  const float* gq     = (const float*)d_in[9];
  const float* bq     = (const float*)d_in[10];
  const float* gk     = (const float*)d_in[11];
  const float* bk     = (const float*)d_in[12];
  const float* gv     = (const float*)d_in[13];
  const float* bv     = (const float*)d_in[14];

  const int B = 8, N = 6273, C = 768;
  const int Nq = 1569, Nk = 393;
  const long nx = (long)B * N * C;       // 38,541,312
  const long nw = (long)C * C;           // 589,824
  const size_t qpoolElems = (size_t)B * 12 * Nq * 64;
  const size_t kpoolElems = (size_t)B * 12 * Nk * 64;
  const size_t oElems = (size_t)B * Nq * C;

  char* ws = (char*)d_ws;
  auto take = [&](size_t bytes) {
    char* p = ws;
    ws += (bytes + 255) & ~(size_t)255;
    return p;
  };
  __bf16* x_bf  = (__bf16*)take((size_t)nx * 2);
  __bf16* wq_bf = (__bf16*)take((size_t)nw * 2);
  __bf16* wk_bf = (__bf16*)take((size_t)nw * 2);
  __bf16* wv_bf = (__bf16*)take((size_t)nw * 2);
  __bf16* wp_bf = (__bf16*)take((size_t)nw * 2);
  float*  qproj = (float*)take((size_t)nx * 4);
  float*  kproj = (float*)take((size_t)nx * 4);
  float*  vproj = (float*)take((size_t)nx * 4);
  __bf16* qpool = (__bf16*)take(qpoolElems * 2);
  __bf16* kpool = (__bf16*)take(kpoolElems * 2);
  __bf16* vpool = (__bf16*)take(kpoolElems * 2);
  __bf16* o_bf  = (__bf16*)take(oElems * 2);

  // 0) fp32 -> bf16 for activations and weights (enables async-DMA staging).
  {
    unsigned gx = (unsigned)((nx / 8 + 255) / 256);
    unsigned gw = (unsigned)((nw / 8 + 255) / 256);
    cvt_bf16<<<gx, 256, 0, stream>>>(x, x_bf, nx);
    cvt_bf16<<<gw, 256, 0, stream>>>(Wq, wq_bf, nw);
    cvt_bf16<<<gw, 256, 0, stream>>>(Wk, wk_bf, nw);
    cvt_bf16<<<gw, 256, 0, stream>>>(Wv, wv_bf, nw);
    cvt_bf16<<<gw, 256, 0, stream>>>(Wproj, wp_bf, nw);
  }

  // 1) Q/K/V projections: (B*N x C) @ (C x C)^T, WMMA bf16 + async staging.
  {
    int M = B * N;  // 50184
    dim3 grid((M + 127) / 128, C / 128);
    gemm_bf16_wmma<<<grid, 256, 0, stream>>>(x_bf, wq_bf, nullptr, qproj, M, C, C);
    gemm_bf16_wmma<<<grid, 256, 0, stream>>>(x_bf, wk_bf, nullptr, kproj, M, C, C);
    gemm_bf16_wmma<<<grid, 256, 0, stream>>>(x_bf, wv_bf, nullptr, vproj, M, C, C);
  }

  // 2) Pool + layernorm -> bf16 (B,12,Np,64).
  {
    long tokq = (long)B * 12 * Nq;
    long tokk = (long)B * 12 * Nk;
    pool_norm<<<(unsigned)((tokq + 3) / 4), 128, 0, stream>>>(
        qproj, pqw, gq, bq, qpool, 2, 14, 14);
    pool_norm<<<(unsigned)((tokk + 3) / 4), 128, 0, stream>>>(
        kproj, pkw, gk, bk, kpool, 4, 7, 7);
    pool_norm<<<(unsigned)((tokk + 3) / 4), 128, 0, stream>>>(
        vproj, pvw, gv, bv, vpool, 4, 7, 7);
  }

  // 3) Fused attention (flash-style) -> bf16 (B*Nq, 768).
  {
    dim3 grid((Nq + 63) / 64, B * 12);
    attn_wmma<<<grid, 128, 0, stream>>>(qpool, kpool, vpool, o_bf, 0.125f);
  }

  // 4) Output projection + bias -> d_out fp32.
  {
    int M = B * Nq;  // 12552
    dim3 grid((M + 127) / 128, C / 128);
    gemm_bf16_wmma<<<grid, 256, 0, stream>>>(o_bf, wp_bf, b_proj,
                                             (float*)d_out, M, C, C);
  }
}
